// MIDXSamplerUniform_20547123544756
// MI455X (gfx1250) — matrix operand order repaired and checked by reading, not compiled, not run on hardware
//
#include <hip/hip_runtime.h>
#include <hip/hip_bf16.h>

// ---------------------------------------------------------------------------
// MIDX negative sampler for MI455X (gfx1250, wave32, WMMA).
// Phase 1: r0/r1 logits + row softmax + s0 mixture weights via
//          v_wmma_f32_16x16x32_bf16 (fp32 inputs down-converted to bf16).
// Phase 2: per-row block builds the full 64x64 conditional CDF table in LDS
//          once (amortized over 256 samples), then each sample is two
//          binary searches + L2-resident bucket lookups.
// ---------------------------------------------------------------------------

typedef __attribute__((ext_vector_type(16))) __bf16 v16bf;
typedef __attribute__((ext_vector_type(8)))  float  v8f;

#define BQ   8192
#define DQ   128
#define KC   64
#define NNEG 256

// --- WMMA fragment loaders (contiguous K = 16h + s order in A and B, wave32) --------------------
// A matrix 16x32 bf16: lane L -> M = L&15 ; K-base = (L<16)?0:16
//   VGPR v in 0..7 : K = base + 2v,2v+1
__device__ __forceinline__ v16bf load_A_bf16(const float* src, int ld,
                                             int rowBase, int kBase, int lane) {
  const int m  = lane & 15;
  const int kb = (lane < 16) ? 0 : 16;
  const float* p = src + (size_t)(rowBase + m) * ld + kBase + kb;
  v16bf a;
#pragma unroll
  for (int v = 0; v < 8; ++v) {
    const int k = 2 * v;
    a[2 * v]     = (__bf16)p[k];
    a[2 * v + 1] = (__bf16)p[k + 1];
  }
  return a;
}

// B matrix 32x16 bf16 where B(kd,n) = M[(nBase+n)*ld + kBase+kd]  (i.e. Mᵀ tile)
// lane L -> N = L&15 ; K-base = (L<16)?0:16 ; VGPR v holds K = base+2v, base+2v+1
__device__ __forceinline__ v16bf load_Bt_bf16(const float* M, int ld,
                                              int nBase, int kBase, int lane) {
  const int n  = lane & 15;
  const int kb = (lane < 16) ? 0 : 16;
  const float* p = M + (size_t)(nBase + n) * ld + kBase + kb;
  v16bf b;
#pragma unroll
  for (int v = 0; v < 8; ++v) {
    b[2 * v]     = (__bf16)p[2 * v];
    b[2 * v + 1] = (__bf16)p[2 * v + 1];
  }
  return b;
}

// ---------------------------------------------------------------------------
// Kernel 1: logits + softmax + s0.   Block = 128 threads = 4 waves.
// Each wave owns a 16-row tile; grid.x = BQ/64.
// ---------------------------------------------------------------------------
__global__ __launch_bounds__(128) void midx_logits_kernel(
    const float* __restrict__ query, const float* __restrict__ c0,
    const float* __restrict__ c1, const float* __restrict__ wkk,
    float* __restrict__ r0g, float* __restrict__ r1g,
    float* __restrict__ r1sg, float* __restrict__ s0g) {
  __shared__ float l0[4][16][KC];   // r0 then r0s
  __shared__ float l1[4][16][KC];   // r1 then r1s
  const int lane    = threadIdx.x & 31;
  const int w       = threadIdx.x >> 5;
  const int rowBase = blockIdx.x * 64 + w * 16;
  const int n       = lane & 15;
  const int moff    = (lane < 16) ? 0 : 8;

  // A fragments: q0 = query[:,0:64], q1 = query[:,64:128], each split K=0..31/32..63
  const v16bf a0lo = load_A_bf16(query, DQ, rowBase, 0,  lane);
  const v16bf a0hi = load_A_bf16(query, DQ, rowBase, 32, lane);
  const v16bf a1lo = load_A_bf16(query, DQ, rowBase, 64, lane);
  const v16bf a1hi = load_A_bf16(query, DQ, rowBase, 96, lane);

#pragma unroll
  for (int t = 0; t < 4; ++t) {               // 4 N-tiles of 16 clusters
    v8f acc0 = {};
    v8f acc1 = {};
    v16bf b = load_Bt_bf16(c0, KC, t * 16, 0, lane);
    acc0 = __builtin_amdgcn_wmma_f32_16x16x32_bf16(false, a0lo, false, b,
                                                   (short)0, acc0, false, false);
    b = load_Bt_bf16(c0, KC, t * 16, 32, lane);
    acc0 = __builtin_amdgcn_wmma_f32_16x16x32_bf16(false, a0hi, false, b,
                                                   (short)0, acc0, false, false);
    b = load_Bt_bf16(c1, KC, t * 16, 0, lane);
    acc1 = __builtin_amdgcn_wmma_f32_16x16x32_bf16(false, a1lo, false, b,
                                                   (short)0, acc1, false, false);
    b = load_Bt_bf16(c1, KC, t * 16, 32, lane);
    acc1 = __builtin_amdgcn_wmma_f32_16x16x32_bf16(false, a1hi, false, b,
                                                   (short)0, acc1, false, false);
#pragma unroll
    for (int v = 0; v < 8; ++v) {
      const int m   = v + moff;
      const int col = t * 16 + n;
      l0[w][m][col] = acc0[v];
      l1[w][m][col] = acc1[v];
      r0g[(size_t)(rowBase + m) * KC + col] = acc0[v];
      r1g[(size_t)(rowBase + m) * KC + col] = acc1[v];
    }
  }
  __syncthreads();

  // Row softmax over K=64 (lanes 0..15, one row each). LDS in-order per wave.
  if (lane < 16) {
    const int m = lane;
    float mx0 = -1e30f, mx1 = -1e30f;
    for (int k = 0; k < KC; ++k) {
      mx0 = fmaxf(mx0, l0[w][m][k]);
      mx1 = fmaxf(mx1, l1[w][m][k]);
    }
    float sum0 = 0.f, sum1 = 0.f;
    for (int k = 0; k < KC; ++k) {
      const float e0 = __expf(l0[w][m][k] - mx0);
      const float e1 = __expf(l1[w][m][k] - mx1);
      l0[w][m][k] = e0;  sum0 += e0;
      l1[w][m][k] = e1;  sum1 += e1;
    }
    const float i0 = 1.f / sum0, i1 = 1.f / sum1;
    for (int k = 0; k < KC; ++k) {
      l0[w][m][k] *= i0;                       // r0s
      const float v1 = l1[w][m][k] * i1;       // r1s
      l1[w][m][k] = v1;
      r1sg[(size_t)(rowBase + m) * KC + k] = v1;
    }
  }
  __syncthreads();

  // s0 = (r1s @ wkkᵀ) * r0s        (A from LDS r1s, B = wkkᵀ tiles from L2)
  const float* r1s_l = &l1[w][0][0];
  const v16bf aslo = load_A_bf16(r1s_l, KC, 0, 0,  lane);
  const v16bf ashi = load_A_bf16(r1s_l, KC, 0, 32, lane);
#pragma unroll
  for (int t = 0; t < 4; ++t) {
    v8f acc = {};
    v16bf b = load_Bt_bf16(wkk, KC, t * 16, 0, lane);
    acc = __builtin_amdgcn_wmma_f32_16x16x32_bf16(false, aslo, false, b,
                                                  (short)0, acc, false, false);
    b = load_Bt_bf16(wkk, KC, t * 16, 32, lane);
    acc = __builtin_amdgcn_wmma_f32_16x16x32_bf16(false, ashi, false, b,
                                                  (short)0, acc, false, false);
#pragma unroll
    for (int v = 0; v < 8; ++v) {
      const int m   = v + moff;
      const int col = t * 16 + n;
      s0g[(size_t)(rowBase + m) * KC + col] = acc[v] * l0[w][m][col];
    }
  }
}

// ---------------------------------------------------------------------------
// Kernel 2: categorical sampling.  Block = 256 threads = one query row.
// Builds the full conditional CDF table cdf1[k0][k] in LDS once per block
// (16.25 KB, padded stride 65 so equal-depth binary-search probes with
// different k0 hit distinct LDS banks), then every sample is two lower-bound
// binary searches (12 LDS probes) plus L2-resident indptr/indices lookups.
// ---------------------------------------------------------------------------
__device__ __forceinline__ unsigned long long splitmix64(unsigned long long x) {
  x += 0x9E3779B97F4A7C15ull;
  unsigned long long z = x;
  z = (z ^ (z >> 30)) * 0xBF58476D1CE4E5B9ull;
  z = (z ^ (z >> 27)) * 0x94D049BB133111EBull;
  return z ^ (z >> 31);
}

__global__ __launch_bounds__(256) void midx_sample_kernel(
    const float* __restrict__ r0g, const float* __restrict__ r1g,
    const float* __restrict__ r1sg, const float* __restrict__ s0g,
    const float* __restrict__ wkk, const int* __restrict__ indices,
    const int* __restrict__ indptr, int nIdx,
    float* __restrict__ out_items, float* __restrict__ out_p) {
  __shared__ float cdf0[KC];
  __shared__ float sr0[KC];
  __shared__ float sr1[KC];
  __shared__ float sr1s[KC];
  __shared__ float cdf1[KC][KC + 1];   // +1 pad: bank = (k0 + k) % 64
  const int b   = blockIdx.x;
  const int tid = threadIdx.x;

  if (tid < KC) {
    cdf0[tid] = s0g[(size_t)b * KC + tid];
    sr0[tid]  = r0g[(size_t)b * KC + tid];
    sr1[tid]  = r1g[(size_t)b * KC + tid];
    sr1s[tid] = r1sg[(size_t)b * KC + tid];
  }
  __syncthreads();
  if (tid == 0) {                 // inclusive CDF over 64 entries (trivial)
    float run = 0.f;
    for (int k = 0; k < KC; ++k) { run += cdf0[k]; cdf0[k] = run; }
  }
  // Conditional CDF table: one builder thread per k0 row, wkk row from L2.
  if (tid < KC) {
    const float* wrow = wkk + (size_t)tid * KC;
    float run = 0.f;
#pragma unroll 4
    for (int k = 0; k < KC; ++k) {
      run += wrow[k] * sr1s[k];
      cdf1[tid][k] = run;
    }
  }
  __syncthreads();

  const unsigned long long gid = (unsigned long long)b * NNEG + tid;
  const unsigned long long h1  = splitmix64(gid * 0x2545F4914F6CDD1Dull + 0x42ull);
  const unsigned long long h2  = splitmix64(h1 ^ 0x6A09E667F3BCC908ull);
  const float u0 = (float)(h1 >> 40) * (1.0f / 16777216.0f);
  const float u1 = (float)(h1 & 0xFFFFFFull) * (1.0f / 16777216.0f);
  const float u2 = (float)(h2 >> 40) * (1.0f / 16777216.0f);

  // k0 ~ Cat(s0[b,:]) : lower-bound binary search on inclusive CDF (6 probes)
  const float target0 = u0 * cdf0[KC - 1];
  int k0 = 0;
#pragma unroll
  for (int s = 32; s >= 1; s >>= 1) {
    const int t = k0 + s;
    if (t < KC && cdf0[t - 1] < target0) k0 = t;
  }

  // k1 ~ Cat(wkk[k0,:] * r1s[b,:]) : binary search on precomputed CDF row
  const float target1 = u1 * cdf1[k0][KC - 1];
  int k1 = 0;
#pragma unroll
  for (int s = 32; s >= 1; s >>= 1) {
    const int t = k1 + s;
    if (t < KC && cdf1[k0][t - 1] < target1) k1 = t;
  }

  // Uniform item inside bucket k01
  const int k01  = k0 * KC + k1;
  const int base = indptr[k01];
  const int cnt  = indptr[k01 + 1] - base;
  int ii = (int)((float)cnt * u2);
  if (ii >= cnt) ii = (cnt > 0) ? cnt - 1 : 0;
  int pos = base + ii;
  if (pos >= nIdx) pos = nIdx - 1;
  if (pos < 0) pos = 0;

  out_items[gid] = (float)(indices[pos] + 1);
  out_p[gid]     = sr0[k0] + sr1[k1];
}

// ---------------------------------------------------------------------------
extern "C" void kernel_launch(void* const* d_in, const int* in_sizes, int n_in,
                              void* d_out, int out_size, void* d_ws, size_t ws_size,
                              hipStream_t stream) {
  const float* query   = (const float*)d_in[0];  // [B, D]
  const float* c0      = (const float*)d_in[1];  // [K, D/2]
  const float* c1      = (const float*)d_in[2];  // [K, D/2]
  const float* wkk     = (const float*)d_in[3];  // [K, K]
  const int*   indices = (const int*)d_in[4];    // [N]
  const int*   indptr  = (const int*)d_in[5];    // [K*K+1]
  const int    nIdx    = in_sizes[4];

  float* ws   = (float*)d_ws;
  float* r0g  = ws;                               // [B, K]
  float* r1g  = r0g  + (size_t)BQ * KC;           // [B, K]
  float* r1sg = r1g  + (size_t)BQ * KC;           // [B, K]
  float* s0g  = r1sg + (size_t)BQ * KC;           // [B, K]

  float* out_items = (float*)d_out;                         // [B, NNEG]
  float* out_p     = out_items + (size_t)BQ * NNEG;         // [B, NNEG]

  hipLaunchKernelGGL(midx_logits_kernel, dim3(BQ / 64), dim3(128), 0, stream,
                     query, c0, c1, wkk, r0g, r1g, r1sg, s0g);
  hipLaunchKernelGGL(midx_sample_kernel, dim3(BQ), dim3(NNEG), 0, stream,
                     r0g, r1g, r1sg, s0g, wkk, indices, indptr, nIdx,
                     out_items, out_p);
}
